// DifferentialAttention_32847909879885
// MI455X (gfx1250) — matrix operand order repaired and checked by reading, not compiled, not run on hardware
//
#include <hip/hip_runtime.h>

typedef _Float16 half_t;
typedef __attribute__((ext_vector_type(16))) _Float16 v16h;
typedef __attribute__((ext_vector_type(8)))  float    v8f;

// Load a 16-half WMMA fragment as two 16-byte chunks (ds_load_b128 / global_load_b128).
__device__ inline v16h ldfrag(const half_t* p0, const half_t* p1) {
    union { float4 q[2]; v16h v; } u;
    u.q[0] = *reinterpret_cast<const float4*>(p0);
    u.q[1] = *reinterpret_cast<const float4*>(p1);
    return u.v;
}

// CDNA5 async copy: global -> LDS, 16B per lane, tracked by ASYNCcnt.
// Generic pointers to __shared__ carry the LDS byte offset in their low 32 bits
// (ISA 10.2).
__device__ inline void async_copy_b128(void* lds_ptr, const void* gptr) {
    unsigned lds_off = (unsigned)(unsigned long long)lds_ptr;
    unsigned long long ga = (unsigned long long)gptr;
    asm volatile("global_load_async_to_lds_b128 %0, %1, off"
                 :: "v"(lds_off), "v"(ga) : "memory");
}
__device__ inline void wait_async0() {
    asm volatile("s_wait_asynccnt 0x0" ::: "memory");
}

// ---------------------------------------------------------------------------
// Elementwise f32 -> f16
// ---------------------------------------------------------------------------
__global__ void da_cvt_f16(const float* __restrict__ in, half_t* __restrict__ out, int n) {
    for (int i = blockIdx.x * blockDim.x + threadIdx.x; i < n; i += gridDim.x * blockDim.x)
        out[i] = (half_t)in[i];
}

// ---------------------------------------------------------------------------
// f16 WMMA GEMM: C[M,N] = alpha * A[M,K] x B[K,N] (+ bias)
// 128 threads = 4 waves; block tile 128x64; K-step 32.
// Wave w computes rows [w*32, w*32+32) x 64 cols = 2x4 wmma tiles.
// A tile staged via async-to-LDS; B tile transposed via manual ds stores.
// ---------------------------------------------------------------------------
__global__ __launch_bounds__(128) void da_gemm_f16(
    const half_t* __restrict__ A, const half_t* __restrict__ B,
    half_t* __restrict__ Ch, float* __restrict__ Cf,
    const float* __restrict__ bias, int M, int N, int K, float alpha)
{
    __shared__ half_t As[128 * 40];  // A tile [row][k], stride 40 halves (80B, 16B-aligned)
    __shared__ half_t Bt[64 * 40];   // B tile transposed [col][k]

    const int tid  = threadIdx.x;
    const int lane = tid & 31;
    const int w    = tid >> 5;
    const int c15  = lane & 15;
    const int hi   = lane >> 4;
    const int n0   = blockIdx.x * 64;
    const int m0   = blockIdx.y * 128;

    v8f acc[2][4] = {};

    for (int kc = 0; kc < K; kc += 32) {
        __syncthreads();
        // A: 128 rows x 32 halves = 512 x 16B chunks, async straight into LDS.
        for (int i = 0; i < 4; ++i) {
            int c = tid + i * 128;
            int row = c >> 2, off = (c & 3) * 8;
            async_copy_b128(&As[row * 40 + off],
                            &A[(size_t)(m0 + row) * K + kc + off]);
        }
        // B: 32 k-rows x 64 cols, transposed into Bt[col][k].
        for (int i = 0; i < 2; ++i) {
            int c = tid + i * 128;
            int k = c >> 3, n8 = (c & 7) * 8;
            union { float4 q; half_t h[8]; } u;
            u.q = *reinterpret_cast<const float4*>(&B[(size_t)(kc + k) * N + n0 + n8]);
            for (int j = 0; j < 8; ++j) Bt[(n8 + j) * 40 + k] = u.h[j];
        }
        wait_async0();
        __syncthreads();

        // A fragments for the wave's two 16-row slabs.
        v16h a[2];
        for (int mt = 0; mt < 2; ++mt) {
            const half_t* ap = &As[(w * 32 + mt * 16 + c15) * 40 + hi * 8];
            a[mt] = ldfrag(ap, ap + 16);
        }
        for (int nt = 0; nt < 4; ++nt) {
            const half_t* bp = &Bt[(nt * 16 + c15) * 40 + hi * 8];
            v16h b = ldfrag(bp, bp + 16);
            for (int mt = 0; mt < 2; ++mt)
                acc[mt][nt] = __builtin_amdgcn_wmma_f32_16x16x32_f16(
                    false, a[mt], false, b, (short)0, acc[mt][nt], false, false);
        }
    }

    const int rbase = hi * 8;
    for (int mt = 0; mt < 2; ++mt)
        for (int nt = 0; nt < 4; ++nt) {
            int col = n0 + nt * 16 + c15;
            float bv = bias ? bias[col] : 0.0f;
            for (int r = 0; r < 8; ++r) {
                int row = m0 + w * 32 + mt * 16 + rbase + r;
                float v = acc[mt][nt][r] * alpha + bv;
                if (Ch) Ch[(size_t)row * N + col] = (half_t)v;
                else    Cf[(size_t)row * N + col] = v;
            }
        }
}

// ---------------------------------------------------------------------------
// Flash attention, one (64-query tile, b*H+h, branch) per block.
// 128 threads = 4 waves; each wave owns 16 query rows. Br=Bc=64, D=64.
// Scale 1/sqrt(D) already folded into Q. K tile staged via async-to-LDS.
// ---------------------------------------------------------------------------
__global__ __launch_bounds__(128) void da_flash_attn(
    const half_t* __restrict__ Qh, const half_t* __restrict__ Kh,
    const half_t* __restrict__ Vh, float* __restrict__ O1, float* __restrict__ O2,
    int N, int ldqk, int ldv)
{
    __shared__ half_t Ks[64 * 72];       // K tile [key][d], stride 72 halves (144B)
    __shared__ half_t Vt[64 * 72];       // V tile transposed [d][key]
    __shared__ half_t Ps[4 * 16 * 72];   // per-wave P tile [row][key]

    const int tid  = threadIdx.x;
    const int lane = tid & 31;
    const int w    = tid >> 5;
    const int c15  = lane & 15;
    const int hi   = lane >> 4;
    const int qt   = blockIdx.x;
    const int bh   = blockIdx.y;
    const int e    = blockIdx.z;
    const int b    = bh / 12, h = bh % 12;
    const int qkcol = (e * 12 + h) * 64;
    const int vcol  = h * 64;
    const size_t rowbase = (size_t)b * N;
    float* O = e ? O2 : O1;

    // Q fragments held in registers for whole kernel (16 rows x 64 d per wave).
    const int qrow = qt * 64 + w * 16 + c15;
    const half_t* qp = &Qh[(rowbase + qrow) * (size_t)ldqk + qkcol];
    v16h qf[2];
    for (int kc = 0; kc < 2; ++kc)
        qf[kc] = ldfrag(qp + kc * 32 + hi * 8, qp + kc * 32 + hi * 8 + 16);

    v8f oacc[4] = {};
    float mrun[8], lrun[8];
    for (int r = 0; r < 8; ++r) { mrun[r] = -1e30f; lrun[r] = 0.0f; }

    for (int j = 0; j < N; j += 64) {
        __syncthreads();
        // Stage tiles: K row-major (async), V transposed (manual).
        for (int i = 0; i < 4; ++i) {
            int c = tid + i * 128;
            int key = c >> 3, d8 = (c & 7) * 8;
            async_copy_b128(&Ks[key * 72 + d8],
                            &Kh[(rowbase + j + key) * (size_t)ldqk + qkcol + d8]);
            union { float4 q; half_t hh[8]; } u;
            u.q = *reinterpret_cast<const float4*>(&Vh[(rowbase + j + key) * (size_t)ldv + vcol + d8]);
            for (int t = 0; t < 8; ++t) Vt[(d8 + t) * 72 + key] = u.hh[t];
        }
        // Hint the next tile toward the caches while we compute this one.
        if (j + 64 < N) {
            int pk = tid >> 1, pc = (tid & 1) * 32;
            __builtin_prefetch(&Kh[(rowbase + j + 64 + pk) * (size_t)ldqk + qkcol + pc], 0, 1);
            __builtin_prefetch(&Vh[(rowbase + j + 64 + pk) * (size_t)ldv + vcol + pc], 0, 1);
        }
        wait_async0();
        __syncthreads();

        // S = Q x K^T  (K^T B-fragment: fixed key, contiguous d run in Ks).
        v8f s[4] = {};
        for (int kc = 0; kc < 2; ++kc)
            for (int nt = 0; nt < 4; ++nt) {
                const half_t* kp = &Ks[(nt * 16 + c15) * 72 + kc * 32 + hi * 8];
                v16h kf = ldfrag(kp, kp + 16);
                s[nt] = __builtin_amdgcn_wmma_f32_16x16x32_f16(
                    false, qf[kc], false, kf, (short)0, s[nt], false, false);
            }

        // Online softmax: rows live across lanes of one 16-lane half.
        float corr[8];
        for (int r = 0; r < 8; ++r) {
            float v = fmaxf(fmaxf(s[0][r], s[1][r]), fmaxf(s[2][r], s[3][r]));
            for (int off = 1; off < 16; off <<= 1) v = fmaxf(v, __shfl_xor(v, off, 32));
            float mnew = fmaxf(mrun[r], v);
            corr[r] = __expf(mrun[r] - mnew);
            mrun[r] = mnew;
        }
        for (int r = 0; r < 8; ++r) {
            float rs = 0.0f;
            for (int nt = 0; nt < 4; ++nt) {
                float p = __expf(s[nt][r] - mrun[r]);
                rs += p;
                Ps[(w * 16 + hi * 8 + r) * 72 + nt * 16 + c15] = (half_t)p;
            }
            for (int off = 1; off < 16; off <<= 1) rs += __shfl_xor(rs, off, 32);
            lrun[r] = lrun[r] * corr[r] + rs;
            for (int nt = 0; nt < 4; ++nt) oacc[nt][r] *= corr[r];
        }
        __syncthreads();   // P written (C-layout) -> read back as A fragments

        // O += P x V  (V^T B-fragment: fixed d, contiguous key run in Vt).
        for (int kc = 0; kc < 2; ++kc) {
            const half_t* pp = &Ps[(w * 16 + c15) * 72 + kc * 32 + hi * 8];
            v16h pf = ldfrag(pp, pp + 16);
            for (int nt = 0; nt < 4; ++nt) {
                const half_t* vp = &Vt[(nt * 16 + c15) * 72 + kc * 32 + hi * 8];
                v16h vf = ldfrag(vp, vp + 16);
                oacc[nt] = __builtin_amdgcn_wmma_f32_16x16x32_f16(
                    false, pf, false, vf, (short)0, oacc[nt], false, false);
            }
        }
    }

    // Normalize and write O[b*N + row, h*64 + d] (f32, [B*N, C] layout).
    for (int r = 0; r < 8; ++r) {
        float inv = 1.0f / lrun[r];
        int orow = qt * 64 + w * 16 + hi * 8 + r;
        for (int nt = 0; nt < 4; ++nt)
            O[(rowbase + orow) * 768 + vcol + nt * 16 + c15] = oacc[nt][r] * inv;
    }
}

// ---------------------------------------------------------------------------
// out_h = O1 - exp(lambda_p[h]) * O2   (f16 for final GEMM)
// ---------------------------------------------------------------------------
__global__ void da_combine(const float* __restrict__ O1, const float* __restrict__ O2,
                           const float* __restrict__ lam_p, half_t* __restrict__ out,
                           int total, int C)
{
    for (int i = blockIdx.x * blockDim.x + threadIdx.x; i < total; i += gridDim.x * blockDim.x) {
        int h = (i % C) >> 6;               // D = 64
        float lam = __expf(lam_p[h]);
        out[i] = (half_t)(O1[i] - lam * O2[i]);
    }
}

// ---------------------------------------------------------------------------
extern "C" void kernel_launch(void* const* d_in, const int* in_sizes, int n_in,
                              void* d_out, int out_size, void* d_ws, size_t ws_size,
                              hipStream_t stream) {
    const float* x    = (const float*)d_in[0];
    const float* Wq   = (const float*)d_in[1];
    const float* Wk   = (const float*)d_in[2];
    const float* Wv   = (const float*)d_in[3];
    const float* lamp = (const float*)d_in[4];
    const float* Wo   = (const float*)d_in[5];
    const float* bo   = (const float*)d_in[6];
    float* out = (float*)d_out;

    const int B = 2, N = 2048, C = 768, H = 12, C2 = 2 * C, M = B * N;

    char* ws = (char*)d_ws;
    size_t off = 0;
    auto carve = [&](size_t bytes) -> void* {
        void* p = ws + off;
        off = (off + bytes + 255) & ~(size_t)255;
        return p;
    };
    half_t* xh    = (half_t*)carve((size_t)M * C  * sizeof(half_t));
    half_t* Wqh   = (half_t*)carve((size_t)C * C2 * sizeof(half_t));
    half_t* Wkh   = (half_t*)carve((size_t)C * C2 * sizeof(half_t));
    half_t* Wvh   = (half_t*)carve((size_t)C * C  * sizeof(half_t));
    half_t* Woh   = (half_t*)carve((size_t)C * C  * sizeof(half_t));
    half_t* Qh    = (half_t*)carve((size_t)M * C2 * sizeof(half_t));
    half_t* Kh    = (half_t*)carve((size_t)M * C2 * sizeof(half_t));
    half_t* Vh    = (half_t*)carve((size_t)M * C  * sizeof(half_t));
    float*  O1    = (float*) carve((size_t)M * C  * sizeof(float));
    float*  O2    = (float*) carve((size_t)M * C  * sizeof(float));
    half_t* attnh = (half_t*)carve((size_t)M * C  * sizeof(half_t));

    // 1) f32 -> f16 conversions
    da_cvt_f16<<<1024, 256, 0, stream>>>(x,  xh,  M * C);
    da_cvt_f16<<<1024, 256, 0, stream>>>(Wq, Wqh, C * C2);
    da_cvt_f16<<<1024, 256, 0, stream>>>(Wk, Wkh, C * C2);
    da_cvt_f16<<<512,  256, 0, stream>>>(Wv, Wvh, C * C);
    da_cvt_f16<<<512,  256, 0, stream>>>(Wo, Woh, C * C);

    // 2) Projections (scale 1/sqrt(D)=0.125 folded into Q)
    da_gemm_f16<<<dim3(C2 / 64, M / 128), 128, 0, stream>>>(
        xh, Wqh, Qh, nullptr, nullptr, M, C2, C, 0.125f);
    da_gemm_f16<<<dim3(C2 / 64, M / 128), 128, 0, stream>>>(
        xh, Wkh, Kh, nullptr, nullptr, M, C2, C, 1.0f);
    da_gemm_f16<<<dim3(C / 64, M / 128), 128, 0, stream>>>(
        xh, Wvh, Vh, nullptr, nullptr, M, C, C, 1.0f);

    // 3) Flash attention: both branches, all (b,h), all query tiles
    da_flash_attn<<<dim3(N / 64, B * H, 2), 128, 0, stream>>>(
        Qh, Kh, Vh, O1, O2, N, C2, C);

    // 4) Differential combine -> f16
    da_combine<<<1024, 256, 0, stream>>>(O1, O2, lamp, attnh, M * C, C);

    // 5) Output projection with bias -> f32 d_out
    da_gemm_f16<<<dim3(C / 64, M / 128), 128, 0, stream>>>(
        attnh, Woh, nullptr, out, bo, M, C, C, 1.0f);
}
